// LPCNet_91061896609827
// MI455X (gfx1250) — matrix operand order repaired and checked by reading, not compiled, not run on hardware
//
#include <hip/hip_runtime.h>
#include <hip/hip_bf16.h>
#include <math.h>

// ---------- types ----------
typedef __attribute__((ext_vector_type(16))) __bf16 v16bf;
typedef __attribute__((ext_vector_type(8)))  float  v8f;
typedef __attribute__((ext_vector_type(4)))  unsigned int u32x4;
struct BF32 { u32x4 a, b; };   // 32 bytes == v16bf

// Opaque pointer: blocks loop-invariant-code-motion of loads through it.
#define OPAQUE_PTR(p) asm volatile("" : "+s"(p))

// Explicit global-address-space views (restores global_load after OPAQUE_PTR
// breaks addrspace inference and would otherwise yield flat_load).
typedef const __attribute__((address_space(1))) u32x4*          gp_u32x4;
typedef const __attribute__((address_space(1))) float*          gp_f32;

__device__ __forceinline__ float gload(const float* p, size_t idx) {
  return ((gp_f32)(unsigned long long)p)[idx];
}

__device__ __forceinline__ unsigned short f2bf(float f) {
  unsigned int u = __float_as_uint(f);
  unsigned int r = (u + 0x7FFFu + ((u >> 16) & 1u)) >> 16;  // RNE
  return (unsigned short)r;
}
__device__ __forceinline__ float bf2f(unsigned short h) {
  return __uint_as_float(((unsigned int)h) << 16);
}
__device__ __forceinline__ float sigm(float x) { return 1.0f / (1.0f + __expf(-x)); }

__device__ __forceinline__ v8f zero8() {
  v8f z = {0.f,0.f,0.f,0.f,0.f,0.f,0.f,0.f};
  return z;
}

// A fragment (16-bit, 16x32, wave32): lane L holds row M = m0 + L%16.
// Element e -> K = kt*32 + 8*(L/16) + (e<8 ? e : e+8)  => two contiguous 8-elem runs
// at K0 and K0+16.  Source is row-major [rows][stride] bf16 in LDS.
__device__ __forceinline__ v16bf loadA(const unsigned short* base, int stride,
                                       int m0, int kt, int lane) {
  int lh  = lane >> 4;
  int row = m0 + (lane & 15);
  int K0  = kt * 32 + 8 * lh;
  const u32x4* p = (const u32x4*)(base + row * stride + K0);
  BF32 t; t.a = p[0]; t.b = p[2];            // +16 bf16 elements = +2 u32x4
  return __builtin_bit_cast(v16bf, t);
}

// B fragment from pre-swizzled global tiles: [kt][nt][lane][16 bf16] contiguous.
// Explicit addrspace(1) => global_load_b128 with scalar base.
__device__ __forceinline__ v16bf loadB(const unsigned short* Bsw, int NT,
                                       int kt, int nt, int lane) {
  gp_u32x4 p = (gp_u32x4)(unsigned long long)
               (Bsw + ((size_t)(kt * NT + nt) * 32 + lane) * 16);
  BF32 t; t.a = p[0]; t.b = p[1];
  return __builtin_bit_cast(v16bf, t);
}

__device__ __forceinline__ v8f wmma_bf16(v16bf a, v16bf b, v8f c) {
  return __builtin_amdgcn_wmma_f32_16x16x32_bf16(false, a, false, b, (short)0, c,
                                                 false, false);
}

// ---------- preprocessing kernels ----------
// Y[r,c] = tanh(X[r,:] @ W[:,c] + bias[c]),  r in [0,1024)
__global__ void k_dense_tanh(const float* __restrict__ X, const float* __restrict__ W,
                             const float* __restrict__ bias, float* __restrict__ Y,
                             int Kdim, int N) {
  int idx = blockIdx.x * blockDim.x + threadIdx.x;
  if (idx >= 1024 * N) return;
  int r = idx / N, c = idx % N;
  float a = bias[c];
  for (int i = 0; i < Kdim; ++i) a += X[r * Kdim + i] * W[i * N + c];
  Y[idx] = tanhf(a);
}

// gxa[r,n] = ba[n] + cond[r,:128] @ Wxa[0:128, n]   (time-invariant part of x@Wxa)
__global__ void k_gxa(const float* __restrict__ cond, const float* __restrict__ Wxa,
                      const float* __restrict__ ba, float* __restrict__ gxa) {
  int r = blockIdx.x;
  for (int c = threadIdx.x; c < 1152; c += blockDim.x) {
    float a = ba[c];
    for (int i = 0; i < 128; ++i) a += cond[r * 128 + i] * Wxa[i * 1152 + c];
    gxa[(size_t)r * 1152 + c] = a;
  }
}

// Convert f32 [srcK, N] row-major weight to bf16 WMMA-B tiles [KT][NT][32][16],
// zero-padding K up to Kd (Kd multiple of 32, N multiple of 16).
__global__ void k_swz(const float* __restrict__ src, unsigned short* __restrict__ dst,
                      int Kd, int N, int srcK) {
  int idx = blockIdx.x * blockDim.x + threadIdx.x;
  if (idx >= Kd * N) return;
  int k = idx / N, n = idx % N;
  int kt = k >> 5, kl = k & 31, nt = n >> 4, nlo = n & 15;
  int NT = N >> 4;
  int lane = nlo + 16 * ((kl >> 3) & 1), e = (kl & 7) + ((kl >> 4) << 3);
  float v = (k < srcK) ? src[k * N + n] : 0.0f;
  dst[((size_t)(kt * NT + nt) * 32 + lane) * 16 + e] = f2bf(v);
}

// ---------- main persistent autoregressive kernel ----------
// grid = 32 blocks (32 rows each), block = 512 threads = 16 waves.
__global__ __launch_bounds__(512, 1)
void lpcnet_main(const float* __restrict__ lpc_all, const float* __restrict__ gumbel,
                 const float* __restrict__ gxa_base, const unsigned short* __restrict__ whaB,
                 const unsigned short* __restrict__ wxbB, const unsigned short* __restrict__ whbB,
                 const unsigned short* __restrict__ w1B, const unsigned short* __restrict__ w2B,
                 const float* __restrict__ Wxa, const float* __restrict__ bb,
                 const float* __restrict__ b1, const float* __restrict__ g1,
                 const float* __restrict__ b2, const float* __restrict__ g2,
                 float* __restrict__ out) {
  __shared__ __align__(16) unsigned short sh_ha[32 * 384];  // GRU-A state, bf16
  __shared__ __align__(16) unsigned short sh_lg[32 * 256];  // logits, bf16
  __shared__ __align__(16) unsigned short sh_hb[32 * 32];   // GRU-B state, bf16, K padded
  __shared__ __align__(16) float sh_gxb[32 * 48];
  __shared__ __align__(16) float sh_ghb[32 * 48];
  __shared__ __align__(16) float sh_s[32 * 16];             // prev_s history
  __shared__ __align__(16) float sh_lpc[32 * 16];
  __shared__ float sh_p[32], sh_last[32], sh_e[32];

  const int tid  = threadIdx.x;
  const int lane = tid & 31;
  const int w    = tid >> 5;
  const int lh   = lane >> 4;
  const int nl   = lane & 15;
  const int rowBase = blockIdx.x * 32;

  // init state
  for (int i = tid; i < 32 * 384; i += 512) sh_ha[i] = 0;
  for (int i = tid; i < 32 * 32;  i += 512) sh_hb[i] = 0;
  for (int i = tid; i < 32 * 16;  i += 512) {
    sh_s[i] = 0.0f;
    sh_lpc[i] = lpc_all[rowBase * 16 + i];
  }
  if (tid < 32) sh_e[tid] = 0.0f;
  __syncthreads();

  // wave roles: mh = M-tile (rows mh*16..+15), hg = h-column group (3 h-tiles)
  const int mh = w >> 3;
  const int hg = w & 7;

  const float inv_log256 = 1.0f / logf(256.0f);

  for (int t = 0; t < 160; ++t) {
    // Per-iteration opaque copies: stop LICM from hoisting 100+ loads out of the
    // time loop (which previously spilled every B fragment to scratch).
    const float*          gxa_t = gxa_base;
    const float*          wxa_t = Wxa;
    const unsigned short* whb_t = whbB;
    const unsigned short* w1_t  = w1B;
    const unsigned short* w2_t  = w2B;
    const float*          b1_t  = b1;
    const float*          g1_t  = g1;
    const float*          b2_t  = b2;
    const float*          g2_t  = g2;
    asm volatile("" : "+s"(gxa_t), "+s"(wxa_t), "+s"(whb_t), "+s"(w1_t), "+s"(w2_t));
    asm volatile("" : "+s"(b1_t), "+s"(g1_t), "+s"(b2_t), "+s"(g2_t));

    // ---- phase A: LPC predict + mu-law quantize (one lane per row) ----
    if (tid < 32) {
      float d = 0.f;
#pragma unroll
      for (int i = 0; i < 16; ++i) d += sh_lpc[tid * 16 + i] * sh_s[tid * 16 + i];
      float xc = fminf(fmaxf(d, -1.f), 1.f);
      float y  = copysignf(log1pf(255.f * fabsf(xc)) * inv_log256, xc);
      float p  = fminf(fmaxf(floorf((y + 1.f) * 128.f), 0.f), 255.f);
      sh_p[tid]    = p;
      sh_last[tid] = sh_s[tid * 16 + 15];
    }

    // ---- GRU-A big GEMM: gh = ha @ Wha  (per wave: 3 h-tiles x 3 gates) ----
    v8f acc[3][3];
#pragma unroll
    for (int i = 0; i < 3; ++i)
#pragma unroll
      for (int g = 0; g < 3; ++g) acc[i][g] = zero8();

    for (int kt = 0; kt < 12; ++kt) {
      // Opaque per-kt: keep B loads inside the kt loop (bounded live range).
      const unsigned short* wha_kt = whaB;
      OPAQUE_PTR(wha_kt);
      v16bf a = loadA(sh_ha, 384, mh * 16, kt, lane);
      // Issue all 9 B-fragment loads first (one s_clause, partial waits),
      // then the 9 WMMAs: overlaps L2 streaming with matrix math.
      v16bf bfr[3][3];
#pragma unroll
      for (int i = 0; i < 3; ++i)
#pragma unroll
        for (int g = 0; g < 3; ++g)
          bfr[i][g] = loadB(wha_kt, 72, kt, g * 24 + hg * 3 + i, lane);
#pragma unroll
      for (int i = 0; i < 3; ++i)
#pragma unroll
        for (int g = 0; g < 3; ++g)
          acc[i][g] = wmma_bf16(a, bfr[i][g], acc[i][g]);
    }
    __syncthreads();  // p/last visible; all reads of sh_ha finished

    // ---- GRU-A gates (register math; r,z,n co-located per wave) ----
    {
      float pv[8], lv[8], ev[8];
#pragma unroll
      for (int v = 0; v < 8; ++v) {
        int r = mh * 16 + v + 8 * lh;
        pv[v] = sh_p[r]; lv[v] = sh_last[r]; ev[v] = sh_e[r];
      }
#pragma unroll
      for (int i = 0; i < 3; ++i) {
        int hcol = (hg * 3 + i) * 16 + nl;
        // rank-1 input weights (rows 128=p, 129=last, 130=e of Wxa), L2-hit loads
        float wpv[3], wlv[3], wev[3];
#pragma unroll
        for (int g = 0; g < 3; ++g) {
          int col = g * 384 + hcol;
          wpv[g] = gload(wxa_t, 128 * 1152 + col);
          wlv[g] = gload(wxa_t, 129 * 1152 + col);
          wev[g] = gload(wxa_t, 130 * 1152 + col);
        }
#pragma unroll
        for (int v = 0; v < 8; ++v) {
          int r  = mh * 16 + v + 8 * lh;
          size_t rg = (size_t)(rowBase + r) * 1152;
          float gx0 = gload(gxa_t, rg + 0 * 384 + hcol) + pv[v]*wpv[0] + lv[v]*wlv[0] + ev[v]*wev[0];
          float gx1 = gload(gxa_t, rg + 1 * 384 + hcol) + pv[v]*wpv[1] + lv[v]*wlv[1] + ev[v]*wev[1];
          float gx2 = gload(gxa_t, rg + 2 * 384 + hcol) + pv[v]*wpv[2] + lv[v]*wlv[2] + ev[v]*wev[2];
          float rr = sigm(gx0 + acc[i][0][v]);
          float zz = sigm(gx1 + acc[i][1][v]);
          float nn = tanhf(gx2 + rr * acc[i][2][v]);
          float ho = bf2f(sh_ha[r * 384 + hcol]);
          sh_ha[r * 384 + hcol] = f2bf((1.f - zz) * nn + zz * ho);
        }
      }
    }
    __syncthreads();  // ha updated

    // ---- GRU-B GEMMs: gxb = ha2@Wxb ; ghb = hb@Whb (K padded 16->32) ----
    if (w < 6) {
      int m2 = (w & 1) * 16, nt = w >> 1;
      v8f a2 = zero8();
      for (int kt = 0; kt < 12; ++kt) {
        const unsigned short* wxb_kt = wxbB;
        OPAQUE_PTR(wxb_kt);
        v16bf af = loadA(sh_ha, 384, m2, kt, lane);
        a2 = wmma_bf16(af, loadB(wxb_kt, 3, kt, nt, lane), a2);
      }
#pragma unroll
      for (int v = 0; v < 8; ++v)
        sh_gxb[(m2 + v + 8 * lh) * 48 + nt * 16 + nl] = a2[v];
    } else if (w < 12) {
      int w2 = w - 6;
      int m2 = (w2 & 1) * 16, nt = w2 >> 1;
      v16bf af = loadA(sh_hb, 32, m2, 0, lane);
      v8f a2 = wmma_bf16(af, loadB(whb_t, 3, 0, nt, lane), zero8());
#pragma unroll
      for (int v = 0; v < 8; ++v)
        sh_ghb[(m2 + v + 8 * lh) * 48 + nt * 16 + nl] = a2[v];
    }
    __syncthreads();

    // ---- GRU-B gates: one thread per (row, h-col) ----
    {
      const float* bb_t = bb;
      OPAQUE_PTR(bb_t);
      int r = tid >> 4, hc = tid & 15;
      float rr = sigm(sh_gxb[r*48 + hc]      + gload(bb_t, hc)      + sh_ghb[r*48 + hc]);
      float zz = sigm(sh_gxb[r*48 + 16 + hc] + gload(bb_t, 16 + hc) + sh_ghb[r*48 + 16 + hc]);
      float nn = tanhf(sh_gxb[r*48 + 32 + hc] + gload(bb_t, 32 + hc) + rr * sh_ghb[r*48 + 32 + hc]);
      float ho = bf2f(sh_hb[r * 32 + hc]);
      sh_hb[r * 32 + hc] = f2bf((1.f - zz) * nn + zz * ho);
    }
    __syncthreads();

    // ---- dual FC: logits = tanh(hb2@W1+b1)*g1 + tanh(hb2@W2+b2)*g2 ----
    {
      v16bf af = loadA(sh_hb, 32, mh * 16, 0, lane);
#pragma unroll
      for (int jj = 0; jj < 2; ++jj) {
        int j = hg + jj * 8;
        int col = j * 16 + nl;
        float bc1 = gload(b1_t, col), gc1 = gload(g1_t, col);
        float bc2 = gload(b2_t, col), gc2 = gload(g2_t, col);
        v8f c1 = wmma_bf16(af, loadB(w1_t, 16, 0, j, lane), zero8());
        v8f c2 = wmma_bf16(af, loadB(w2_t, 16, 0, j, lane), zero8());
#pragma unroll
        for (int v = 0; v < 8; ++v) {
          int r = mh * 16 + v + 8 * lh;
          float lgt = tanhf(c1[v] + bc1) * gc1 + tanhf(c2[v] + bc2) * gc2;
          out[((size_t)(rowBase + r) * 160 + t) * 256 + col] = lgt;
          sh_lg[r * 256 + col] = f2bf(lgt);
        }
      }
    }
    __syncthreads();

    // ---- Gumbel-softmax expected index + state update (16 lanes per row) ----
    {
      int r = tid >> 4, ch = tid & 15;
      const float* gu = gumbel + ((size_t)t * 1024 + rowBase + r) * 256 + ch * 16;
      float vals[16];
      float vmax = -3.0e38f;
#pragma unroll
      for (int i = 0; i < 16; ++i) {
        float g  = -logf(-logf(gu[i]));
        float vv = bf2f(sh_lg[r * 256 + ch * 16 + i]) + g;
        vals[i] = vv;
        vmax = fmaxf(vmax, vv);
      }
#pragma unroll
      for (int off = 8; off >= 1; off >>= 1)
        vmax = fmaxf(vmax, __shfl_xor(vmax, off, 16));
      float s0 = 0.f, s1 = 0.f;
#pragma unroll
      for (int i = 0; i < 16; ++i) {
        float ex = __expf(vals[i] - vmax);
        s0 += ex;
        s1 += ex * (float)(ch * 16 + i);
      }
#pragma unroll
      for (int off = 8; off >= 1; off >>= 1) {
        s0 += __shfl_xor(s0, off, 16);
        s1 += __shfl_xor(s1, off, 16);
      }
      float e = rintf(s1 / s0);             // straight-through forward value
      float s = sh_p[r] + e;
      float old = sh_s[r * 16 + ch];        // wave-synchronous read-before-write
      if (ch > 0) sh_s[r * 16 + ch - 1] = old;
      if (ch == 0) { sh_s[r * 16 + 15] = s; sh_e[r] = e; }
    }
    __syncthreads();
  }
}

// ---------- launcher ----------
extern "C" void kernel_launch(void* const* d_in, const int* in_sizes, int n_in,
                              void* d_out, int out_size, void* d_ws, size_t ws_size,
                              hipStream_t stream) {
  (void)in_sizes; (void)n_in; (void)out_size; (void)ws_size;
  const float* feat = (const float*)d_in[0];
  const float* lpc  = (const float*)d_in[1];
  const float* gum  = (const float*)d_in[2];
  const float* Wf1  = (const float*)d_in[3];
  const float* bf1  = (const float*)d_in[4];
  const float* Wf2  = (const float*)d_in[5];
  const float* bf2  = (const float*)d_in[6];
  const float* Wxa  = (const float*)d_in[7];
  const float* Wha  = (const float*)d_in[8];
  const float* ba   = (const float*)d_in[9];
  const float* Wxb  = (const float*)d_in[10];
  const float* Whb  = (const float*)d_in[11];
  const float* bb   = (const float*)d_in[12];
  const float* W1   = (const float*)d_in[13];
  const float* b1   = (const float*)d_in[14];
  const float* g1   = (const float*)d_in[15];
  const float* W2   = (const float*)d_in[16];
  const float* b2   = (const float*)d_in[17];
  const float* g2   = (const float*)d_in[18];
  float* out = (float*)d_out;

  char* ws = (char*)d_ws;
  float* cond1 = (float*)(ws);                         // 1024*128 f32 (512 KB)
  float* cond2 = (float*)(ws + (512 << 10));           // 1024*128 f32
  float* gxa   = (float*)(ws + (1 << 20));             // 1024*1152 f32 (4.5 MB)
  unsigned short* whaB = (unsigned short*)(ws + 6 * (1 << 20));              // 884 KB
  unsigned short* wxbB = (unsigned short*)(ws + 7 * (1 << 20));              // 36 KB
  unsigned short* whbB = (unsigned short*)(ws + 7 * (1 << 20) + (64 << 10));
  unsigned short* w1B  = (unsigned short*)(ws + 7 * (1 << 20) + (128 << 10));
  unsigned short* w2B  = (unsigned short*)(ws + 7 * (1 << 20) + (192 << 10));

  // frame-rate network + hoisted conditioning GEMM
  k_dense_tanh<<<(1024 * 128 + 255) / 256, 256, 0, stream>>>(feat,  Wf1, bf1, cond1, 20, 128);
  k_dense_tanh<<<(1024 * 128 + 255) / 256, 256, 0, stream>>>(cond1, Wf2, bf2, cond2, 128, 128);
  k_gxa<<<1024, 256, 0, stream>>>(cond2, Wxa, ba, gxa);
  // weight conversion to bf16 WMMA-B tile layout (K zero-padded where needed)
  k_swz<<<(384 * 1152 + 255) / 256, 256, 0, stream>>>(Wha, whaB, 384, 1152, 384);
  k_swz<<<(384 * 48 + 255) / 256, 256, 0, stream>>>(Wxb, wxbB, 384, 48, 384);
  k_swz<<<(32 * 48 + 255) / 256, 256, 0, stream>>>(Whb, whbB, 32, 48, 16);
  k_swz<<<(32 * 256 + 255) / 256, 256, 0, stream>>>(W1, w1B, 32, 256, 16);
  k_swz<<<(32 * 256 + 255) / 256, 256, 0, stream>>>(W2, w2B, 32, 256, 16);
  // persistent autoregressive kernel: 32 blocks x 32 rows, T=160 internal steps
  lpcnet_main<<<32, 512, 0, stream>>>(lpc, gum, gxa, whaB, wxbB, whbB, w1B, w2B,
                                      Wxa, bb, b1, g1, b2, g2, out);
}
